// InternVisionEncoderLayer_46892452937871
// MI455X (gfx1250) — matrix operand order repaired and checked by reading, not compiled, not run on hardware
//
#include <hip/hip_runtime.h>

// ---------------- problem constants ----------------
#define B_    8
#define N_    1025
#define C_    1024
#define H_    16
#define D_    64
#define FF_   4096
#define M_    (B_ * N_)        // 8200 token rows
#define NPAD  1056             // keys padded so paired V loads stay in bounds
#define NT_   65               // ceil(1025/16)
#define EPSF  1e-6f
#define SCALEF 0.125f          // D^-0.5

// GEMM tiling
#define TM    128
#define TN    128
#define TK    32
#define LPITCH 56              // ushorts; 112B rows: 16B-aligned, conflict-free

// ---------------- WMMA types ----------------
typedef __attribute__((ext_vector_type(16))) __bf16 v16bf;
typedef __attribute__((ext_vector_type(8)))  float  v8f;
typedef __attribute__((ext_vector_type(4)))  int    v4i;

union AFrag { v16bf v; unsigned int u[8]; };
union BFrag { v16bf v; unsigned int u[8]; };
union CFrag { v8f   v; float        f[8]; };

__device__ __forceinline__ v8f wmma_bf16(v16bf a, v16bf b, v8f c) {
  return __builtin_amdgcn_wmma_f32_16x16x32_bf16(false, a, false, b, (short)0, c, false, false);
}

// A-fragment K index for 16x32 bf16 A: lanes 0-15 hold K {0..15},
// lanes 16-31 hold K {16..31}; VGPR v holds the pair (k,k+1).
__device__ __forceinline__ int a_kbase(int v, int half) {
  return 2 * v + 16 * half;
}

__device__ __forceinline__ unsigned short f32_to_bf16(float f) {
  unsigned int u = __float_as_uint(f);
  u += 0x7FFFu + ((u >> 16) & 1u);       // round-to-nearest-even
  return (unsigned short)(u >> 16);
}

// ---------------- CDNA5 async global->LDS copy (guarded) ----------------
#define AS1G __attribute__((address_space(1)))
#define AS3L __attribute__((address_space(3)))

#if defined(__has_builtin)
#if __has_builtin(__builtin_amdgcn_global_load_async_to_lds_b128) && \
    __has_builtin(__builtin_amdgcn_s_wait_asynccnt)
#define USE_ASYNC_LDS 1
#endif
#endif
#ifndef USE_ASYNC_LDS
#define USE_ASYNC_LDS 0
#endif

__device__ __forceinline__ void copy16_g2l(const unsigned short* src,
                                           unsigned short* dst_lds) {
#if USE_ASYNC_LDS
  __builtin_amdgcn_global_load_async_to_lds_b128(
      (AS1G v4i*)(src), (AS3L v4i*)(dst_lds), 0, 0);
#else
  *(uint4*)dst_lds = *(const uint4*)src;
#endif
}

__device__ __forceinline__ void async_wait_all() {
#if USE_ASYNC_LDS
  __builtin_amdgcn_s_wait_asynccnt(0);
#endif
}

// ---------------- elementwise f32 -> bf16 weight convert ----------------
__global__ void cvt_bf16_kernel(const float* __restrict__ in,
                                unsigned short* __restrict__ out, int n) {
  for (int i = blockIdx.x * blockDim.x + threadIdx.x; i < n;
       i += gridDim.x * blockDim.x)
    out[i] = f32_to_bf16(in[i]);
}

// ---------------- LayerNorm over C=1024, f32 in -> bf16 out ----------------
__global__ __launch_bounds__(256)
void ln_bf16_kernel(const float* __restrict__ x, const float* __restrict__ w,
                    const float* __restrict__ b, unsigned short* __restrict__ out) {
  __shared__ float r1[256];
  __shared__ float r2[256];
  int row = blockIdx.x, tid = threadIdx.x;
  const float* xr = x + (size_t)row * C_;
  float v[4], s = 0.f, s2 = 0.f;
#pragma unroll
  for (int i = 0; i < 4; ++i) {
    v[i] = xr[tid + i * 256];
    s += v[i]; s2 += v[i] * v[i];
  }
  r1[tid] = s; r2[tid] = s2;
  __syncthreads();
  for (int st = 128; st > 0; st >>= 1) {
    if (tid < st) { r1[tid] += r1[tid + st]; r2[tid] += r2[tid + st]; }
    __syncthreads();
  }
  float mean = r1[0] * (1.f / C_);
  float var  = r2[0] * (1.f / C_) - mean * mean;
  float rs   = rsqrtf(var + EPSF);
#pragma unroll
  for (int i = 0; i < 4; ++i) {
    int c = tid + i * 256;
    out[(size_t)row * C_ + c] = f32_to_bf16((v[i] - mean) * rs * w[c] + b[c]);
  }
}

// ------- QKV post: RMSNorm(Q)*scale, RMSNorm(K), V transposed+padded -------
__global__ __launch_bounds__(256)
void qkv_pack_kernel(const float* __restrict__ qkv, const float* __restrict__ qw,
                     const float* __restrict__ kw,
                     unsigned short* __restrict__ qh, unsigned short* __restrict__ kh,
                     unsigned short* __restrict__ vt) {
  int blk = blockIdx.x, tid = threadIdx.x;
  int b = blk / NPAD, n = blk % NPAD;
  if (n >= N_) {                 // zero the V pad region so paired loads are safe
#pragma unroll
    for (int i = 0; i < 4; ++i) {
      int c = tid + i * 256, h = c >> 6, d = c & 63;
      vt[(((size_t)(b * H_ + h)) * D_ + d) * NPAD + n] = 0;
    }
    return;
  }
  __shared__ float r1[256];
  __shared__ float r2[256];
  const float* rowp = qkv + ((size_t)(b * N_ + n)) * (3 * C_);
  float q[4], k[4], vv[4], sq = 0.f, sk = 0.f;
#pragma unroll
  for (int i = 0; i < 4; ++i) {
    int c = tid + i * 256;
    q[i]  = rowp[c];
    k[i]  = rowp[C_ + c];
    vv[i] = rowp[2 * C_ + c];
    sq += q[i] * q[i]; sk += k[i] * k[i];
  }
  r1[tid] = sq; r2[tid] = sk;
  __syncthreads();
  for (int st = 128; st > 0; st >>= 1) {
    if (tid < st) { r1[tid] += r1[tid + st]; r2[tid] += r2[tid + st]; }
    __syncthreads();
  }
  float rq = rsqrtf(r1[0] * (1.f / C_) + EPSF);
  float rk = rsqrtf(r2[0] * (1.f / C_) + EPSF);
#pragma unroll
  for (int i = 0; i < 4; ++i) {
    int c = tid + i * 256, h = c >> 6, d = c & 63;
    size_t bh = (size_t)(b * H_ + h);
    qh[(bh * N_ + n) * D_ + d] = f32_to_bf16(q[i] * rq * qw[c] * SCALEF);
    kh[(bh * N_ + n) * D_ + d] = f32_to_bf16(k[i] * rk * kw[c]);
    vt[(bh * D_ + d) * NPAD + n] = f32_to_bf16(vv[i]);
  }
}

// ---------------- tiled WMMA GEMM: out[M,Nout] = A[M,K] * W[Nout,K]^T ------
// Double-buffered LDS, async global->LDS when available.
// mode 0: f32 out (+bias)   mode 1: bias+GELU -> bf16   mode 2: resid+acc*ls
__global__ __launch_bounds__(256)
void gemm_bf16_kernel(const unsigned short* __restrict__ A,
                      const unsigned short* __restrict__ W,
                      float* __restrict__ Cf, unsigned short* __restrict__ Cb,
                      const float* __restrict__ bias, const float* __restrict__ ls,
                      const float* __restrict__ resid,
                      int Mtot, int K, int Nout, int mode) {
  __shared__ alignas(16) unsigned short As[2][TM][LPITCH];
  __shared__ alignas(16) unsigned short Bs[2][TN][LPITCH];
  int tid = threadIdx.x, lane = tid & 31, wv = tid >> 5;
  int half = lane >> 4, lm = lane & 15;
  int m0 = blockIdx.y * TM, n0 = blockIdx.x * TN;
  int wm = (wv >> 1) * 32, wn = (wv & 1) * 64;

  CFrag acc[2][4];
#pragma unroll
  for (int i = 0; i < 2; ++i)
#pragma unroll
    for (int j = 0; j < 4; ++j)
#pragma unroll
      for (int r = 0; r < 8; ++r) acc[i][j].f[r] = 0.f;

  // stage copy: 128x32 A tile + 128x32 W tile, 16B per slot, 4 slots/row.
  auto issue_stage = [&](int k0, int buf) {
#pragma unroll
    for (int i = 0; i < 2; ++i) {
      int slot = tid + i * 256;
      int row = slot >> 2, col = (slot & 3) * 8;
      int gm = m0 + row;
      if (gm >= Mtot) gm = Mtot - 1;            // clamp (masked at epilogue)
      copy16_g2l(A + (size_t)gm * K + k0 + col, &As[buf][row][col]);
    }
#pragma unroll
    for (int i = 0; i < 2; ++i) {
      int slot = tid + i * 256;
      int row = slot >> 2, col = (slot & 3) * 8;
      copy16_g2l(W + (size_t)(n0 + row) * K + k0 + col, &Bs[buf][row][col]);
    }
  };

  int nstage = K / TK;
  issue_stage(0, 0);

  for (int s = 0; s < nstage; ++s) {
    int buf = s & 1;
    async_wait_all();
    __syncthreads();
    if (s + 1 < nstage) issue_stage((s + 1) * TK, buf ^ 1);

    AFrag a[2]; BFrag b[4];
#pragma unroll
    for (int i = 0; i < 2; ++i) {
      int row = wm + i * 16 + lm;
#pragma unroll
      for (int v = 0; v < 8; ++v) {
        int k = a_kbase(v, half);
        a[i].u[v] = *(const unsigned int*)&As[buf][row][k];
      }
    }
#pragma unroll
    for (int j = 0; j < 4; ++j) {
      int col = wn + j * 16 + lm;
#pragma unroll
      for (int v = 0; v < 8; ++v) {
        int k = 2 * v + 16 * half;
        b[j].u[v] = *(const unsigned int*)&Bs[buf][col][k];
      }
    }
#pragma unroll
    for (int i = 0; i < 2; ++i)
#pragma unroll
      for (int j = 0; j < 4; ++j)
        acc[i][j].v = wmma_bf16(a[i].v, b[j].v, acc[i][j].v);
  }

#pragma unroll
  for (int i = 0; i < 2; ++i)
#pragma unroll
    for (int j = 0; j < 4; ++j)
#pragma unroll
      for (int r = 0; r < 8; ++r) {
        int gm = m0 + wm + i * 16 + r + 8 * half;
        int gn = n0 + wn + j * 16 + lm;
        if (gm < Mtot) {
          float v = acc[i][j].f[r];
          if (bias) v += bias[gn];
          size_t idx = (size_t)gm * Nout + gn;
          if (mode == 0) {
            Cf[idx] = v;
          } else if (mode == 1) {
            float g = 0.5f * v * (1.f + erff(v * 0.70710678118f));
            Cb[idx] = f32_to_bf16(g);
          } else {
            Cf[idx] = resid[idx] + v * ls[gn];
          }
        }
      }
}

// ---------------- flash attention: one block per (b,h), 8 waves ------------
__global__ __launch_bounds__(256)
void attn_kernel(const unsigned short* __restrict__ qh,
                 const unsigned short* __restrict__ kh,
                 const unsigned short* __restrict__ vt,
                 unsigned short* __restrict__ obuf) {
  __shared__ alignas(16) unsigned short Ps[8][16][36];  // per-wave P staging
  int tid = threadIdx.x, lane = tid & 31, wv = tid >> 5;
  int half = lane >> 4, lm = lane & 15;
  int bh = blockIdx.x, b = bh >> 4, h = bh & 15;
  const unsigned short* qhb = qh + (size_t)bh * N_ * D_;
  const unsigned short* khb = kh + (size_t)bh * N_ * D_;
  const unsigned short* vtb = vt + (size_t)bh * D_ * NPAD;
  volatile unsigned short* psS = &Ps[wv][0][0];
  volatile const unsigned int* psL = (volatile const unsigned int*)&Ps[wv][0][0];

  for (int qt = wv; qt < NT_; qt += 8) {
    AFrag aq[2];
    int qrow = qt * 16 + lm;
    bool qok = qrow < N_;
#pragma unroll
    for (int kk = 0; kk < 2; ++kk)
#pragma unroll
      for (int v = 0; v < 8; ++v) {
        int d = kk * 32 + a_kbase(v, half);
        aq[kk].u[v] = qok ? *(const unsigned int*)(qhb + (size_t)qrow * D_ + d) : 0u;
      }

    float mrow[8], lrow[8];
    CFrag o[4];
#pragma unroll
    for (int r = 0; r < 8; ++r) {
      mrow[r] = -1e30f; lrow[r] = 0.f;
      o[0].f[r] = 0.f; o[1].f[r] = 0.f; o[2].f[r] = 0.f; o[3].f[r] = 0.f;
    }

    for (int kt = 0; kt < NT_; ++kt) {
      int key = kt * 16 + lm;
      bool kok = key < N_;
      BFrag bk[2];
#pragma unroll
      for (int kk = 0; kk < 2; ++kk)
#pragma unroll
        for (int v = 0; v < 8; ++v) {
          int d = 2 * v + 16 * half + 32 * kk;
          bk[kk].u[v] = kok ? *(const unsigned int*)(khb + (size_t)key * D_ + d) : 0u;
        }
      CFrag s;
#pragma unroll
      for (int r = 0; r < 8; ++r) s.f[r] = 0.f;
      s.v = wmma_bf16(aq[0].v, bk[0].v, s.v);
      s.v = wmma_bf16(aq[1].v, bk[1].v, s.v);

#pragma unroll
      for (int r = 0; r < 8; ++r) {
        float sv = kok ? s.f[r] : -1e30f;
        float mx = sv;
        mx = fmaxf(mx, __shfl_xor(mx, 1, 32));
        mx = fmaxf(mx, __shfl_xor(mx, 2, 32));
        mx = fmaxf(mx, __shfl_xor(mx, 4, 32));
        mx = fmaxf(mx, __shfl_xor(mx, 8, 32));
        float mn = fmaxf(mrow[r], mx);
        float alpha = __expf(mrow[r] - mn);
        float p = __expf(sv - mn);
        float ps = p;
        ps += __shfl_xor(ps, 1, 32);
        ps += __shfl_xor(ps, 2, 32);
        ps += __shfl_xor(ps, 4, 32);
        ps += __shfl_xor(ps, 8, 32);
        lrow[r] = lrow[r] * alpha + ps;
        mrow[r] = mn;
        o[0].f[r] *= alpha; o[1].f[r] *= alpha;
        o[2].f[r] *= alpha; o[3].f[r] *= alpha;
        psS[(r + 8 * half) * 36 + lm]      = f32_to_bf16(p);
        psS[(r + 8 * half) * 36 + lm + 16] = 0;      // zero K=16..31 pad
      }

      AFrag ap;                                       // P as 16x32 A-fragment
#pragma unroll
      for (int v = 0; v < 8; ++v) {
        int k = a_kbase(v, half);
        ap.u[v] = psL[(lm * 36 + k) >> 1];
      }
      BFrag bv[4];
#pragma unroll
      for (int j = 0; j < 4; ++j) {
        int d = j * 16 + lm;
#pragma unroll
        for (int v = 0; v < 8; ++v) {
          int k2 = 2 * v + 16 * half;
          int key2 = kt * 16 + k2;                    // pad rows are zero
          bv[j].u[v] = *(const unsigned int*)(vtb + (size_t)d * NPAD + key2);
        }
      }
#pragma unroll
      for (int j = 0; j < 4; ++j) o[j].v = wmma_bf16(ap.v, bv[j].v, o[j].v);
    }

#pragma unroll
    for (int j = 0; j < 4; ++j)
#pragma unroll
      for (int r = 0; r < 8; ++r) {
        int row = qt * 16 + r + 8 * half;
        if (row < N_) {
          float val = o[j].f[r] / lrow[r];
          obuf[((size_t)(b * N_ + row)) * C_ + h * D_ + j * 16 + lm] =
              f32_to_bf16(val);
        }
      }
  }
}

// ---------------- host orchestration ----------------
extern "C" void kernel_launch(void* const* d_in, const int* in_sizes, int n_in,
                              void* d_out, int out_size, void* d_ws, size_t ws_size,
                              hipStream_t stream) {
  const float* x       = (const float*)d_in[0];
  const float* w_qkv   = (const float*)d_in[1];
  const float* w_proj  = (const float*)d_in[2];
  const float* b_proj  = (const float*)d_in[3];
  const float* q_norm  = (const float*)d_in[4];
  const float* k_norm  = (const float*)d_in[5];
  const float* ln1_w   = (const float*)d_in[6];
  const float* ln1_b   = (const float*)d_in[7];
  const float* ln2_w   = (const float*)d_in[8];
  const float* ln2_b   = (const float*)d_in[9];
  const float* w_fc1   = (const float*)d_in[10];
  const float* b_fc1   = (const float*)d_in[11];
  const float* w_fc2   = (const float*)d_in[12];
  const float* b_fc2   = (const float*)d_in[13];
  const float* ls1     = (const float*)d_in[14];
  const float* ls2     = (const float*)d_in[15];

  char* ws = (char*)d_ws;
  size_t off = 0;
  auto alloc = [&](size_t bytes) -> char* {
    char* p = ws + off;
    off += (bytes + 255) & ~(size_t)255;
    return p;
  };

  unsigned short* h1_bf   = (unsigned short*)alloc((size_t)M_ * C_ * 2);
  unsigned short* wqkv_bf = (unsigned short*)alloc((size_t)3 * C_ * C_ * 2);
  unsigned short* wproj_bf= (unsigned short*)alloc((size_t)C_ * C_ * 2);
  unsigned short* wfc1_bf = (unsigned short*)alloc((size_t)FF_ * C_ * 2);
  unsigned short* wfc2_bf = (unsigned short*)alloc((size_t)C_ * FF_ * 2);
  float*          qkv_f   = (float*)alloc((size_t)M_ * 3 * C_ * 4);
  unsigned short* qh      = (unsigned short*)alloc((size_t)B_ * H_ * N_ * D_ * 2);
  unsigned short* kh      = (unsigned short*)alloc((size_t)B_ * H_ * N_ * D_ * 2);
  unsigned short* vt      = (unsigned short*)alloc((size_t)B_ * H_ * D_ * NPAD * 2);
  unsigned short* obuf    = (unsigned short*)alloc((size_t)M_ * C_ * 2);
  float*          x2      = (float*)alloc((size_t)M_ * C_ * 4);
  unsigned short* h2_bf   = (unsigned short*)alloc((size_t)M_ * C_ * 2);
  unsigned short* fc1_bf  = (unsigned short*)alloc((size_t)M_ * FF_ * 2);
  (void)ws_size; (void)in_sizes; (void)n_in; (void)out_size;

  // 1) weight converts
  cvt_bf16_kernel<<<2048, 256, 0, stream>>>(w_qkv,  wqkv_bf, 3 * C_ * C_);
  cvt_bf16_kernel<<<2048, 256, 0, stream>>>(w_proj, wproj_bf, C_ * C_);
  cvt_bf16_kernel<<<2048, 256, 0, stream>>>(w_fc1,  wfc1_bf, FF_ * C_);
  cvt_bf16_kernel<<<2048, 256, 0, stream>>>(w_fc2,  wfc2_bf, C_ * FF_);

  // 2) LN1 -> bf16
  ln_bf16_kernel<<<M_, 256, 0, stream>>>(x, ln1_w, ln1_b, h1_bf);

  // 3) QKV GEMM (f32 out)
  gemm_bf16_kernel<<<dim3(3 * C_ / TN, (M_ + TM - 1) / TM), 256, 0, stream>>>(
      h1_bf, wqkv_bf, qkv_f, nullptr, nullptr, nullptr, nullptr,
      M_, C_, 3 * C_, 0);

  // 4) RMSNorm Q/K, pack V transposed
  qkv_pack_kernel<<<B_ * NPAD, 256, 0, stream>>>(qkv_f, q_norm, k_norm, qh, kh, vt);

  // 5) attention
  attn_kernel<<<B_ * H_, 256, 0, stream>>>(qh, kh, vt, obuf);

  // 6) proj + bias + layer-scale residual -> x2
  gemm_bf16_kernel<<<dim3(C_ / TN, (M_ + TM - 1) / TM), 256, 0, stream>>>(
      obuf, wproj_bf, x2, nullptr, b_proj, ls1, x, M_, C_, C_, 2);

  // 7) LN2 -> bf16
  ln_bf16_kernel<<<M_, 256, 0, stream>>>(x2, ln2_w, ln2_b, h2_bf);

  // 8) FC1 + bias + GELU -> bf16
  gemm_bf16_kernel<<<dim3(FF_ / TN, (M_ + TM - 1) / TM), 256, 0, stream>>>(
      h2_bf, wfc1_bf, nullptr, fc1_bf, b_fc1, nullptr, nullptr,
      M_, C_, FF_, 1);

  // 9) FC2 + bias + layer-scale residual -> out
  gemm_bf16_kernel<<<dim3(C_ / TN, (M_ + TM - 1) / TM), 256, 0, stream>>>(
      fc1_bf, wfc2_bf, (float*)d_out, nullptr, b_fc2, ls2, x2,
      M_, FF_, C_, 2);
}
